// Conv_72954314490495
// MI455X (gfx1250) — matrix operand order repaired and checked by reading, not compiled, not run on hardware
//
#include <hip/hip_runtime.h>

typedef __attribute__((ext_vector_type(16))) __bf16 v16bf;
typedef __attribute__((ext_vector_type(8)))  float  v8f;

#define DFEAT 128
#define DOUT  128

// ---------- helpers ----------

// order-preserving f32 -> u32 encoding (monotonic: a<b  <=>  enc(a)<enc(b))
__device__ __forceinline__ unsigned enc_f32(float f) {
  unsigned u = __float_as_uint(f);
  return (u & 0x80000000u) ? ~u : (u | 0x80000000u);
}
__device__ __forceinline__ float dec_f32(unsigned e) {
  unsigned b = (e & 0x80000000u) ? (e ^ 0x80000000u) : ~e;
  return __uint_as_float(b);
}

// load two contiguous 8-float chunks and convert to a bf16 x16 fragment
__device__ __forceinline__ v16bf cvt16(const float* __restrict__ p0,
                                       const float* __restrict__ p1) {
  float4 a0 = *(const float4*)(p0);
  float4 a1 = *(const float4*)(p0 + 4);
  float4 b0 = *(const float4*)(p1);
  float4 b1 = *(const float4*)(p1 + 4);
  v16bf r;
  r[0]  = (__bf16)a0.x; r[1]  = (__bf16)a0.y; r[2]  = (__bf16)a0.z; r[3]  = (__bf16)a0.w;
  r[4]  = (__bf16)a1.x; r[5]  = (__bf16)a1.y; r[6]  = (__bf16)a1.z; r[7]  = (__bf16)a1.w;
  r[8]  = (__bf16)b0.x; r[9]  = (__bf16)b0.y; r[10] = (__bf16)b0.z; r[11] = (__bf16)b0.w;
  r[12] = (__bf16)b1.x; r[13] = (__bf16)b1.y; r[14] = (__bf16)b1.z; r[15] = (__bf16)b1.w;
  return r;
}

// ---------- kernel 1: h = feat @ W_pool^T + b_pool ----------
// one 16x16 output tile per wave; 4 waves (128 threads) per block
__global__ void gemm_pool_kernel(const float* __restrict__ feat,
                                 const float* __restrict__ Wp,   // [128,128] row-major
                                 const float* __restrict__ bp,   // [128]
                                 float* __restrict__ h,          // [M,128]
                                 int M) {
  const int wave = threadIdx.x >> 5;
  const int lane = threadIdx.x & 31;
  const int half = lane >> 4;
  const int l16  = lane & 15;

  const int ntn  = DFEAT / 16;                  // 8 N-tiles
  int tile = blockIdx.x * 4 + wave;
  int mt = tile / ntn;
  int nt = tile - mt * ntn;
  if ((mt + 1) * 16 > M) return;                // wave-uniform

  const int rowA = mt * 16 + l16;               // this lane's A row (M)
  const int colB = nt * 16 + l16;               // this lane's output column (N)
  const float* arow = feat + (size_t)rowA * DFEAT;
  const float* wrow = Wp   + (size_t)colB * DFEAT;

  float bias = bp[colB];
  v8f c = { bias, bias, bias, bias, bias, bias, bias, bias };

  #pragma unroll
  for (int kk = 0; kk < DFEAT; kk += 32) {
    // A 16x32 bf16: lane half selects K groups [8h,8h+8) and [16+8h,16+8h+8)
    v16bf a = cvt16(arow + kk + 8 * half, arow + kk + 16 + 8 * half);
    // B 32x16 bf16: lane half selects K groups [8h,8h+8) and [16+8h,16+8h+8)
    v16bf b = cvt16(wrow + kk + 8 * half, wrow + kk + 16 + 8 * half);
    c = __builtin_amdgcn_wmma_f32_16x16x32_bf16(false, a, false, b,
                                                (short)0, c, false, false);
  }

  // D layout: VGPR r -> row (r + 8*half), column = l16
  float* obase = h + (size_t)(mt * 16) * DFEAT + colB;
  #pragma unroll
  for (int r = 0; r < 8; ++r)
    obase[(size_t)(r + 8 * half) * DFEAT] = c[r];
}

// ---------- kernel 2: per-edge scatter-max (encoded u32 atomics) ----------
__global__ void edge_scatter_max(const float* __restrict__ h,
                                 const float* __restrict__ w,
                                 const int* __restrict__ src,
                                 const int* __restrict__ dst,
                                 unsigned* __restrict__ neigh_enc,
                                 int E) {
  int t = blockIdx.x * blockDim.x + threadIdx.x;
  int e = t >> 5;
  if (e >= E) return;
  int lane = t & 31;

  int s = src[e];
  int d = dst[e];
  float we = w[e];

  float4 hv = *(const float4*)(h + (size_t)s * DFEAT + lane * 4);
  unsigned* out = neigh_enc + (size_t)d * DFEAT + lane * 4;
  atomicMax(out + 0, enc_f32(hv.x * we));
  atomicMax(out + 1, enc_f32(hv.y * we));
  atomicMax(out + 2, enc_f32(hv.z * we));
  atomicMax(out + 3, enc_f32(hv.w * we));
}

// ---------- kernel 3: decode in place, enc==0 (untouched) -> 0.0f ----------
__global__ void decode_kernel(unsigned* __restrict__ p, int n) {
  int i = blockIdx.x * blockDim.x + threadIdx.x;
  if (i >= n) return;
  unsigned e = p[i];
  float f = (e == 0u) ? 0.0f : dec_f32(e);
  p[i] = __float_as_uint(f);
}

// ---------- kernel 4: rst = concat(feat, neigh) @ W_neigh^T + b_neigh ----------
__global__ void gemm_neigh_kernel(const float* __restrict__ feat,
                                  const float* __restrict__ neigh,
                                  const float* __restrict__ Wn,  // [128, 256] row-major
                                  const float* __restrict__ bn,  // [128]
                                  float* __restrict__ out,       // [M, 128]
                                  int M) {
  const int wave = threadIdx.x >> 5;
  const int lane = threadIdx.x & 31;
  const int half = lane >> 4;
  const int l16  = lane & 15;

  const int ntn  = DOUT / 16;                   // 8 N-tiles
  int tile = blockIdx.x * 4 + wave;
  int mt = tile / ntn;
  int nt = tile - mt * ntn;
  if ((mt + 1) * 16 > M) return;                // wave-uniform

  const int rowA = mt * 16 + l16;
  const int colB = nt * 16 + l16;
  const float* frow = feat  + (size_t)rowA * DFEAT;
  const float* nrow = neigh + (size_t)rowA * DFEAT;
  const float* wrow = Wn    + (size_t)colB * (2 * DFEAT);

  float bias = bn[colB];
  v8f c = { bias, bias, bias, bias, bias, bias, bias, bias };

  #pragma unroll
  for (int kk = 0; kk < 2 * DFEAT; kk += 32) {
    const float* arow = (kk < DFEAT) ? (frow + kk) : (nrow + (kk - DFEAT));
    v16bf a = cvt16(arow + 8 * half, arow + 16 + 8 * half);
    v16bf b = cvt16(wrow + kk + 8 * half, wrow + kk + 16 + 8 * half);
    c = __builtin_amdgcn_wmma_f32_16x16x32_bf16(false, a, false, b,
                                                (short)0, c, false, false);
  }

  float* obase = out + (size_t)(mt * 16) * DOUT + colB;
  #pragma unroll
  for (int r = 0; r < 8; ++r)
    obase[(size_t)(r + 8 * half) * DOUT] = c[r];
}

// ---------- launcher ----------
extern "C" void kernel_launch(void* const* d_in, const int* in_sizes, int n_in,
                              void* d_out, int out_size, void* d_ws, size_t ws_size,
                              hipStream_t stream) {
  const float* feat   = (const float*)d_in[0];
  const float* weight = (const float*)d_in[1];
  const int*   src    = (const int*)d_in[2];
  const int*   dst    = (const int*)d_in[3];
  const float* Wp     = (const float*)d_in[4];
  const float* bp     = (const float*)d_in[5];
  const float* Wn     = (const float*)d_in[6];
  const float* bn     = (const float*)d_in[7];
  float* out = (float*)d_out;

  const int M = in_sizes[0] / DFEAT;   // 50000 nodes
  const int E = in_sizes[1];           // 640000 edges

  // workspace: h [M*128 f32] | neigh_enc [M*128 u32] (decoded to f32 in place)
  float*    h         = (float*)d_ws;
  unsigned* neigh_enc = (unsigned*)d_ws + (size_t)M * DFEAT;
  float*    neigh     = (float*)neigh_enc;

  const int total = M * DFEAT;

  // 1) sentinel-init encoded max buffer
  hipMemsetAsync(neigh_enc, 0, (size_t)total * sizeof(unsigned), stream);

  // 2) h = feat @ W_pool^T + b_pool
  const int tiles = (M / 16) * (DFEAT / 16);   // 25000 wave-tiles
  gemm_pool_kernel<<<(tiles + 3) / 4, 128, 0, stream>>>(feat, Wp, bp, h, M);

  // 3) scatter-max over edges (L2-resident h and neigh)
  long long ethreads = (long long)E * 32;
  edge_scatter_max<<<(unsigned)((ethreads + 255) / 256), 256, 0, stream>>>(
      h, weight, src, dst, neigh_enc, E);

  // 4) decode, zero-degree -> 0
  decode_kernel<<<(total + 255) / 256, 256, 0, stream>>>(neigh_enc, total);

  // 5) rst = concat(feat, neigh) @ W_neigh^T + b_neigh
  gemm_neigh_kernel<<<(tiles + 3) / 4, 128, 0, stream>>>(feat, neigh, Wn, bn, out, M);
}
